// dynamicVAE32_78219944395136
// MI455X (gfx1250) — matrix operand order repaired and checked by reading, not compiled, not run on hardware
//
#include <hip/hip_runtime.h>
#include <hip/hip_bf16.h>
#include <math.h>

typedef __attribute__((ext_vector_type(16))) _Float16 v16h;
typedef __attribute__((ext_vector_type(8)))  float    v8f;

#if defined(__HIP_DEVICE_COMPILE__) && __has_builtin(__builtin_amdgcn_global_load_async_to_lds_b128)
#define USE_ASYNC_LDS 1
#else
#define USE_ASYNC_LDS 0
#endif

#if USE_ASYNC_LDS
typedef int i32x4 __attribute__((vector_size(16)));
typedef __attribute__((address_space(1))) i32x4 gi32x4;   // global int4
typedef __attribute__((address_space(3))) i32x4 li32x4;   // LDS int4

__device__ inline gi32x4* as_global_v4(const void* p) {
    return (gi32x4*)const_cast<void*>(p);
}
__device__ inline li32x4* as_lds_v4(void* p) {
    return (li32x4*)p;
}
__device__ inline void wait_async0() {
#if __has_builtin(__builtin_amdgcn_s_wait_asynccnt)
    __builtin_amdgcn_s_wait_asynccnt(0);
#else
    asm volatile("s_wait_asynccnt 0" ::: "memory");
#endif
}
#endif

// ---------------------------------------------------------------- utilities
__global__ void k_f32_to_f16(const float* __restrict__ in, _Float16* __restrict__ out, int n) {
    int i = blockIdx.x * blockDim.x + threadIdx.x;
    if (i < n) out[i] = (_Float16)in[i];
}

// ------------------------------------------------- encoder conv (s=2, p=1, ReLU)
__global__ void k_conv2d_s2_relu(const float* __restrict__ in, const float* __restrict__ w,
                                 const float* __restrict__ bias, float* __restrict__ out,
                                 int N, int Cin, int Cout, int Hin) {
    int Hout = Hin >> 1;
    int total = N * Cout * Hout * Hout;
    int idx = blockIdx.x * blockDim.x + threadIdx.x;
    if (idx >= total) return;
    int ox = idx % Hout; int t = idx / Hout;
    int oy = t % Hout;   t /= Hout;
    int co = t % Cout;   int nb = t / Cout;
    float acc = bias[co];
    const float* inB = in + (size_t)nb * Cin * Hin * Hin;
    const float* wB  = w  + (size_t)co * Cin * 9;
    for (int ci = 0; ci < Cin; ++ci) {
        const float* inC = inB + ci * Hin * Hin;
        const float* wC  = wB + ci * 9;
#pragma unroll
        for (int ky = 0; ky < 3; ++ky) {
            int iy = oy * 2 - 1 + ky;
            if (iy < 0 || iy >= Hin) continue;
#pragma unroll
            for (int kx = 0; kx < 3; ++kx) {
                int ix = ox * 2 - 1 + kx;
                if (ix < 0 || ix >= Hin) continue;
                acc += inC[iy * Hin + ix] * wC[ky * 3 + kx];
            }
        }
    }
    out[idx] = fmaxf(acc, 0.f);
}

// ------------------------------- transposed conv (k=3,s=2,p=1,op=1), ELU/sigmoid
__global__ void k_convT2d_s2(const float* __restrict__ in, const float* __restrict__ w,
                             const float* __restrict__ bias, float* __restrict__ out,
                             int N, int Cin, int Cout, int Hin, int act) {
    int Hout = Hin * 2;
    int total = N * Cout * Hout * Hout;
    int idx = blockIdx.x * blockDim.x + threadIdx.x;
    if (idx >= total) return;
    int ox = idx % Hout; int t = idx / Hout;
    int oy = t % Hout;   t /= Hout;
    int co = t % Cout;   int nb = t / Cout;
    float acc = bias[co];
    const float* inB = in + (size_t)nb * Cin * Hin * Hin;
    for (int ci = 0; ci < Cin; ++ci) {
        const float* inC = inB + ci * Hin * Hin;
        const float* wC  = w + ((size_t)ci * Cout + co) * 9;
#pragma unroll
        for (int ky = 0; ky < 3; ++ky) {
            int ty = oy + 1 - ky;
            if (ty < 0 || (ty & 1)) continue;
            int iy = ty >> 1;
            if (iy >= Hin) continue;
#pragma unroll
            for (int kx = 0; kx < 3; ++kx) {
                int tx = ox + 1 - kx;
                if (tx < 0 || (tx & 1)) continue;
                int ix = tx >> 1;
                if (ix >= Hin) continue;
                acc += inC[iy * Hin + ix] * wC[ky * 3 + kx];
            }
        }
    }
    if (act == 0) out[idx] = (acc > 0.f) ? acc : (expf(acc) - 1.f);   // ELU
    else          out[idx] = 1.f / (1.f + expf(-acc));                // sigmoid
}

// ---------------- WMMA GEMM: C[M,N] = A[M,K](f16) @ W[N,K](f16)^T + bias[N]
// One wave per 16x64 C strip (4 accumulators). grid = (N/64, M/16), block = 32.
// A tile staged through LDS with async-to-LDS copies when the toolchain has them.
__global__ void k_gemm_wmma4(const _Float16* __restrict__ A, const _Float16* __restrict__ W,
                             const float* __restrict__ bias, float* __restrict__ C,
                             int M, int N, int K) {
    int lane = threadIdx.x & 31;
    int g  = lane >> 4;      // half-wave group
    int lr = lane & 15;
    int nBase = blockIdx.x * 64;
    int mTile = blockIdx.y * 16;

    const _Float16* Wr[4];
#pragma unroll
    for (int t = 0; t < 4; ++t)
        Wr[t] = W + (size_t)(nBase + 16 * t + lr) * K;   // B column n = lr of tile t

    v8f acc[4];
#pragma unroll
    for (int t = 0; t < 4; ++t) {
        float bv = bias[nBase + 16 * t + lr];
#pragma unroll
        for (int r = 0; r < 8; ++r) acc[t][r] = bv;
    }

#if USE_ASYNC_LDS
    __shared__ __align__(16) _Float16 As[16 * 32];       // one 16x32 f16 A tile
    int srow = lane >> 1;                                // staging row handled by lane
    int scol = (lane & 1) * 16;                          // 32-byte half-row
    const _Float16* Agrow = A + (size_t)(mTile + srow) * K + scol;
    _Float16* Alrow = As + srow * 32 + scol;
#else
    const _Float16* Arow = A + (size_t)(mTile + lr) * K; // A row m = lr
#endif

    for (int kb = 0; kb < K; kb += 32) {
        v16h a;
#if USE_ASYNC_LDS
        __builtin_amdgcn_global_load_async_to_lds_b128(as_global_v4(Agrow + kb),
                                                       as_lds_v4(Alrow),     0, 0);
        __builtin_amdgcn_global_load_async_to_lds_b128(as_global_v4(Agrow + kb + 8),
                                                       as_lds_v4(Alrow + 8), 0, 0);
        wait_async0();
        const _Float16* afr = As + lr * 32;              // fragment: K = 8g..8g+7, 16+8g..23+8g
#pragma unroll
        for (int e = 0; e < 8; ++e)  a[e]     = afr[8 * g + e];
#pragma unroll
        for (int e = 0; e < 8; ++e)  a[8 + e] = afr[16 + 8 * g + e];
#else
        __builtin_prefetch(Arow + kb + 128, 0, 0);
#pragma unroll
        for (int p = 0; p < 8; ++p) {
            int k0 = kb + ((p < 4) ? (2 * p) : (16 + 2 * (p - 4))) + g * 8;
            a[2 * p]     = Arow[k0];
            a[2 * p + 1] = Arow[k0 + 1];
        }
#endif
#pragma unroll
        for (int t = 0; t < 4; ++t) {
            const _Float16* wp = Wr[t] + kb + g * 8;     // K = 8g..8g+7, 16+8g..23+8g
            v16h b;
#pragma unroll
            for (int e = 0; e < 16; ++e) b[e] = wp[(e < 8) ? e : (e + 8)];
            acc[t] = __builtin_amdgcn_wmma_f32_16x16x32_f16(false, a, false, b,
                                                            (short)0, acc[t], false, false);
        }
    }

#pragma unroll
    for (int t = 0; t < 4; ++t)
#pragma unroll
        for (int r = 0; r < 8; ++r)
            C[(size_t)(mTile + r + g * 8) * N + nBase + 16 * t + lr] = acc[t][r];
}

// ------------- build block-tridiagonal precision matrix Pm = blocks + 0.5*I
__global__ void k_build_pm(const float* __restrict__ Dv, const float* __restrict__ Bv,
                           float* __restrict__ Pm, float* __restrict__ Awork) {
    int idx = blockIdx.x * blockDim.x + threadIdx.x;   // 256*256*256
    if (idx >= (1 << 24)) return;
    int j = idx & 255;
    int i = (idx >> 8) & 255;
    int b = idx >> 16;
    int t1 = i >> 4, p1 = i & 15, t2 = j >> 4, p2 = j & 15;
    const float* Db = Dv + (size_t)b * 4096;
    const float* Bb = Bv + (size_t)b * 3840;
    float v = 0.f;
    if (t1 == t2)          v = Db[(t1 * 16 + p1) * 16 + p2];          // diagonal block
    else if (t1 == t2 + 1) v = Bb[(t2 * 16 + p1) * 16 + p2];          // sub-diagonal
    else if (t2 == t1 + 1) v = Bb[(t1 * 16 + p2) * 16 + p1];          // super (transposed)
    if (i == j) v += 0.5f;                                            // ALPHA * I
    size_t o = ((size_t)b << 16) + i * 256 + j;
    Pm[o] = v;
    Awork[o] = v;
}

// ------------- per-sample 256x256 Cholesky, column broadcast through LDS
__global__ void k_cholesky(float* __restrict__ A) {
    int b = blockIdx.x, tid = threadIdx.x;
    float* Ab = A + ((size_t)b << 16);
    __shared__ float colA[256];
    __shared__ float colL[256];
    for (int k = 0; k < 256; ++k) {
        colA[tid] = Ab[tid * 256 + k];
        __syncthreads();
        float diag = sqrtf(colA[k]);
        float lik;
        if (tid > k)       lik = colA[tid] / diag;
        else if (tid == k) lik = diag;
        else               lik = 0.f;                 // zero upper triangle
        Ab[tid * 256 + k] = lik;
        colL[tid] = lik;
        __syncthreads();
        if (tid > k) {
            float li = lik;
            float* row = Ab + tid * 256;
            for (int j = k + 1; j <= tid; ++j) row[j] -= li * colL[j];
        }
        __syncthreads();
    }
}

// ------------- per-sample lower-triangular inverse: X = L^-1 (thread = column)
__global__ void k_trinv(const float* __restrict__ L, float* __restrict__ X) {
    int b = blockIdx.x, c = threadIdx.x;
    const float* Lb = L + ((size_t)b << 16);
    float* Xb = X + ((size_t)b << 16);
    for (int i = 0; i < c; ++i) Xb[i * 256 + c] = 0.f;  // strict upper = 0
    Xb[c * 256 + c] = 1.f / Lb[c * 256 + c];
    for (int i = c + 1; i < 256; ++i) {
        float s = 0.f;
        const float* Lr = Lb + i * 256;
        for (int j = c; j < i; ++j) s += Lr[j] * Xb[j * 256 + c];
        Xb[i * 256 + c] = -s / Lr[i];
    }
}

// ------------- z = mu + cov @ eps (cov lower-triangular)
__global__ void k_reparam(const float* __restrict__ mu, const float* __restrict__ cov,
                          const float* __restrict__ eps, float* __restrict__ z) {
    int idx = blockIdx.x * blockDim.x + threadIdx.x;   // 256*256
    if (idx >= 65536) return;
    int i = idx & 255, b = idx >> 8;
    const float* cr = cov + ((size_t)b << 16) + i * 256;
    const float* er = eps + b * 256;
    float s = mu[idx];
    for (int j = 0; j <= i; ++j) s += cr[j] * er[j];
    z[idx] = s;
}

// ================================================================ launcher
extern "C" void kernel_launch(void* const* d_in, const int* in_sizes, int n_in,
                              void* d_out, int out_size, void* d_ws, size_t ws_size,
                              hipStream_t stream) {
    (void)in_sizes; (void)n_in; (void)out_size; (void)ws_size;
    const float* x    = (const float*)d_in[0];
    const float* eps  = (const float*)d_in[1];
    const float* w1   = (const float*)d_in[2];  const float* b1 = (const float*)d_in[3];
    const float* w2   = (const float*)d_in[4];  const float* b2 = (const float*)d_in[5];
    const float* w3   = (const float*)d_in[6];  const float* b3 = (const float*)d_in[7];
    const float* w4   = (const float*)d_in[8];  const float* b4 = (const float*)d_in[9];
    const float* w5   = (const float*)d_in[10]; const float* b5 = (const float*)d_in[11];
    const float* fc_mu_w  = (const float*)d_in[12]; const float* fc_mu_b  = (const float*)d_in[13];
    const float* fc_D_w   = (const float*)d_in[14]; const float* fc_D_b   = (const float*)d_in[15];
    const float* fc_B_w   = (const float*)d_in[16]; const float* fc_B_b   = (const float*)d_in[17];
    const float* fc_dec_w = (const float*)d_in[18]; const float* fc_dec_b = (const float*)d_in[19];
    const float* wt5 = (const float*)d_in[20]; const float* bt5 = (const float*)d_in[21];
    const float* wt4 = (const float*)d_in[22]; const float* bt4 = (const float*)d_in[23];
    const float* wt3 = (const float*)d_in[24]; const float* bt3 = (const float*)d_in[25];
    const float* wt2 = (const float*)d_in[26]; const float* bt2 = (const float*)d_in[27];
    const float* wt1 = (const float*)d_in[28]; const float* bt1 = (const float*)d_in[29];

    // output layout: xr | mu | cov | Pm
    float* out = (float*)d_out;
    float* xr  = out;                    // 4096*1024
    float* mu  = out + 4194304;          // 256*256
    float* cov = out + 4259840;          // 256*256*256
    float* Pm  = out + 21037056;         // 256*256*256

    // workspace layout (float slots); bufA doubles as Cholesky work + hd4
    float* ws = (float*)d_ws;
    size_t off = 0;
    float* bufA = ws + off; off += 33554432;  // h1 [4096,32,16,16] / Awork / hd4
    float* bufB = ws + off; off += 8388608;   // h2 [4096,32,8,8]  / hd3
    float* bufC = ws + off; off += 4194304;   // h3 [4096,64,4,4]  / hd2
    float* bufD = ws + off; off += 1048576;   // h4 [4096,64,2,2]  / hd1
    float* h5   = ws + off; off += 262144;    // [4096,64] == h [256,1024]
    _Float16* h16 = (_Float16*)(ws + off); off += 131072;
    float* Dv   = ws + off; off += 1048576;   // [256,4096]
    float* Bv   = ws + off; off += 983040;    // [256,3840]
    float* zf   = ws + off; off += 65536;     // [256,256]
    _Float16* z16 = (_Float16*)(ws + off); off += 32768;
    float* hdec = ws + off; off += 262144;    // [256,1024] == [4096,64,1,1]
    _Float16* wmu16  = (_Float16*)(ws + off); off += 131072;   // 262144 h
    _Float16* wD16   = (_Float16*)(ws + off); off += 2097152;  // 4194304 h
    _Float16* wB16   = (_Float16*)(ws + off); off += 1966080;  // 3932160 h
    _Float16* wdec16 = (_Float16*)(ws + off); off += 131072;   // 262144 h

    const int TPB = 256;
    auto blk = [](int n) { return (n + 255) / 256; };

    // --- FC weights -> f16 ---
    k_f32_to_f16<<<blk(262144),  TPB, 0, stream>>>(fc_mu_w,  wmu16,  262144);
    k_f32_to_f16<<<blk(4194304), TPB, 0, stream>>>(fc_D_w,   wD16,   4194304);
    k_f32_to_f16<<<blk(3932160), TPB, 0, stream>>>(fc_B_w,   wB16,   3932160);
    k_f32_to_f16<<<blk(262144),  TPB, 0, stream>>>(fc_dec_w, wdec16, 262144);

    // --- encoder convs ---
    k_conv2d_s2_relu<<<blk(33554432), TPB, 0, stream>>>(x,    w1, b1, bufA, 4096,  1, 32, 32);
    k_conv2d_s2_relu<<<blk(8388608),  TPB, 0, stream>>>(bufA, w2, b2, bufB, 4096, 32, 32, 16);
    k_conv2d_s2_relu<<<blk(4194304),  TPB, 0, stream>>>(bufB, w3, b3, bufC, 4096, 32, 64,  8);
    k_conv2d_s2_relu<<<blk(1048576),  TPB, 0, stream>>>(bufC, w4, b4, bufD, 4096, 64, 64,  4);
    k_conv2d_s2_relu<<<blk(262144),   TPB, 0, stream>>>(bufD, w5, b5, h5,   4096, 64, 64,  2);
    k_f32_to_f16<<<blk(262144), TPB, 0, stream>>>(h5, h16, 262144);

    // --- FC heads via WMMA: h[256,1024] @ W^T ---
    k_gemm_wmma4<<<dim3(4, 16),  32, 0, stream>>>(h16, wmu16, fc_mu_b, mu, 256,  256, 1024);
    k_gemm_wmma4<<<dim3(64, 16), 32, 0, stream>>>(h16, wD16,  fc_D_b,  Dv, 256, 4096, 1024);
    k_gemm_wmma4<<<dim3(60, 16), 32, 0, stream>>>(h16, wB16,  fc_B_b,  Bv, 256, 3840, 1024);

    // --- precision matrix, Cholesky, inverse, reparametrize ---
    k_build_pm<<<blk(16777216), TPB, 0, stream>>>(Dv, Bv, Pm, bufA);
    k_cholesky<<<256, 256, 0, stream>>>(bufA);
    k_trinv<<<256, 256, 0, stream>>>(bufA, cov);
    k_reparam<<<blk(65536), TPB, 0, stream>>>(mu, cov, eps, zf);
    k_f32_to_f16<<<blk(65536), TPB, 0, stream>>>(zf, z16, 65536);

    // --- decoder FC via WMMA: z[256,256] @ fc_dec_w^T -> [256,1024] ---
    k_gemm_wmma4<<<dim3(16, 16), 32, 0, stream>>>(z16, wdec16, fc_dec_b, hdec, 256, 1024, 256);

    // --- decoder transposed convs (reuse encoder buffers in reverse) ---
    k_convT2d_s2<<<blk(1048576),  TPB, 0, stream>>>(hdec, wt5, bt5, bufD, 4096, 64, 64,  1, 0);
    k_convT2d_s2<<<blk(4194304),  TPB, 0, stream>>>(bufD, wt4, bt4, bufC, 4096, 64, 64,  2, 0);
    k_convT2d_s2<<<blk(8388608),  TPB, 0, stream>>>(bufC, wt3, bt3, bufB, 4096, 64, 32,  4, 0);
    k_convT2d_s2<<<blk(33554432), TPB, 0, stream>>>(bufB, wt2, bt2, bufA, 4096, 32, 32,  8, 0);
    k_convT2d_s2<<<blk(4194304),  TPB, 0, stream>>>(bufA, wt1, bt1, xr,   4096, 32,  1, 16, 1);
}
